// RNNTJoint_52149492908815
// MI455X (gfx1250) — compile-verified, matrix operand order checked by reading, not run on hardware
//
#include <hip/hip_runtime.h>
#include <hip/hip_bf16.h>

// RNNT joint, fused for MI455X (gfx1250, wave32, WMMA, async-LDS).
// Stage 0: pre-swizzle out_w into WMMA B-fragment lane order (bf16) -> ws
// Stage 1: f = enc_out @ enc_w^T + enc_b  (bf16 WMMA, f32 accum) -> ws (f32)
//          g = pred_out @ pred_w^T + pred_b                      -> ws (f32)
// Stage 2: logits = relu(f[t]+g[u]) @ out_w^T + out_b, fully fused:
//          the [B,T,U,H] joint tensor (671 MB) exists only as bf16
//          A-fragments in VGPRs. Staging uses GLOBAL_LOAD_ASYNC_TO_LDS_B128
//          (ASYNCcnt); joint math stays f32 (v_dual_add_f32 + v_max +
//          v_cvt_pk_bf16_f32 — the path this target codegens natively).
// Workspace: (4096 + 1024) * 320 f32 + 30 KB B-frags ~= 6.6 MB.

typedef __bf16 bf16;
typedef __attribute__((ext_vector_type(16))) __bf16 v16bf;
typedef __attribute__((ext_vector_type(8)))  __bf16 v8bf;
typedef __attribute__((ext_vector_type(8)))  float  v8f;
typedef __attribute__((ext_vector_type(4)))  float  f32x4;
typedef int v4i_g __attribute__((vector_size(16)));   // matches builtin param

#define ENC_H   768
#define PRED_H  320
#define JOINT_H 320
#define NUM_C   34
#define B_      8
#define T_      512
#define U_      128

// ---- async global->LDS staging (CDNA5), with portable fallback -------------
#if defined(__has_builtin)
#if __has_builtin(__builtin_amdgcn_global_load_async_to_lds_b128)
#define HAVE_ASYNC_LDS 1
#endif
#endif

#ifdef HAVE_ASYNC_LDS
#define CP16(ldst, gsrc)                                                    \
  __builtin_amdgcn_global_load_async_to_lds_b128(                           \
      (__attribute__((address_space(1))) v4i_g*)(gsrc),                     \
      (__attribute__((address_space(3))) v4i_g*)(ldst), 0, 0)
#if __has_builtin(__builtin_amdgcn_s_wait_asynccnt)
#define WAIT_ASYNC() __builtin_amdgcn_s_wait_asynccnt(0)
#else
#define WAIT_ASYNC() asm volatile("s_wait_asynccnt 0x0" ::: "memory")
#endif
#else
#define CP16(ldst, gsrc) (*(v4i_g*)(ldst) = *(const v4i_g*)(gsrc))
#define WAIT_ASYNC() ((void)0)
#endif

// ---------------------------------------------------------------------------
// Stage 0: pre-swizzle out_w (f32 [34,320]) into the per-lane WMMA B-fragment
// slot order, padded C 34->48 (3 N-tiles), 10 k-steps, bf16. 30 KB, run once.
// Element e of lane L for k-step ks holds B[k,n] = out_w[ct*16 + L%16, k],
// k = ks*32 + 8*(L/16) + e + (e>=8 ? 8 : 0)   (16-bit operand layout).
// ---------------------------------------------------------------------------
__global__ __launch_bounds__(256) void rnnt_prep_bfrag(
    const float* __restrict__ outw, bf16* __restrict__ bws)
{
  for (int i = threadIdx.x; i < 3 * 10 * 32 * 16; i += 256) {
    int e    = i & 15;
    int lane = (i >> 4) & 31;
    int ks   = (i >> 9) % 10;
    int ct   = i / (512 * 10);
    int hi   = lane >> 4, n = lane & 15;
    int c    = ct * 16 + n;
    int k    = ks * 32 + 8 * hi + e + ((e & 8) ? 8 : 0);
    bws[i] = (bf16)((c < NUM_C) ? outw[c * JOINT_H + k] : 0.0f);
  }
}

// ---------------------------------------------------------------------------
// Stage 1: projection GEMM  out[M,320] = in[M,K] @ w[320,K]^T + bias  (f32)
// One wave per 16x16 tile, K swept 32/step with v_wmma_f32_16x16x32_bf16.
// ---------------------------------------------------------------------------
__global__ __launch_bounds__(256) void rnnt_proj_gemm(
    const float* __restrict__ in, const float* __restrict__ w,
    const float* __restrict__ bias, float* __restrict__ out,
    int M, int K)
{
  const int tid  = threadIdx.x;
  const int lane = tid & 31;
  const int wv   = tid >> 5;
  const int gw   = blockIdx.x * 8 + wv;
  const int nHT  = JOINT_H / 16;           // 20 h-tiles
  const int mt   = gw / nHT;
  const int ht   = gw - mt * nHT;
  if (mt * 16 >= M) return;                // wave-uniform exit

  const int lm = lane & 15;
  const int hi = lane >> 4;
  const float* arow = in + (size_t)(mt * 16 + lm) * K;
  const float* brow = w  + (size_t)(ht * 16 + lm) * K;

  v8f acc = {};
  const int nks = K >> 5;
  for (int ks = 0; ks < nks; ++ks) {
    const int b1 = ks * 32 + 8 * hi;       // 32B-aligned f32 runs
    f32x4 a0 = *(const f32x4*)(arow + b1);
    f32x4 a1 = *(const f32x4*)(arow + b1 + 4);
    f32x4 a2 = *(const f32x4*)(arow + b1 + 16);
    f32x4 a3 = *(const f32x4*)(arow + b1 + 20);
    f32x4 w0 = *(const f32x4*)(brow + b1);
    f32x4 w1 = *(const f32x4*)(brow + b1 + 4);
    f32x4 w2 = *(const f32x4*)(brow + b1 + 16);
    f32x4 w3 = *(const f32x4*)(brow + b1 + 20);
    v16bf av, bv;
#pragma unroll
    for (int e = 0; e < 4; ++e) {
      av[e]      = (bf16)a0[e]; av[e + 4]  = (bf16)a1[e];
      av[e + 8]  = (bf16)a2[e]; av[e + 12] = (bf16)a3[e];
      bv[e]      = (bf16)w0[e]; bv[e + 4]  = (bf16)w1[e];
      bv[e + 8]  = (bf16)w2[e]; bv[e + 12] = (bf16)w3[e];
    }
    acc = __builtin_amdgcn_wmma_f32_16x16x32_bf16(
        false, av, false, bv, (short)0, acc, false, false);
  }

  const float bb = bias[ht * 16 + lm];
#pragma unroll
  for (int r = 0; r < 8; ++r) {            // D: M = r + 8*(lane/16), N = lane%16
    out[(size_t)(mt * 16 + r + 8 * hi) * JOINT_H + ht * 16 + lm] = acc[r] + bb;
  }
}

// ---------------------------------------------------------------------------
// Stage 2: fused joint + output projection.
// Workgroup = (b, 16 t's, 64 u's). LDS: f tile f32 (broadcast reads),
// g block f32 (rows padded to 324 f32: stride == 4 dwords mod 64 banks, so
// the 16 lanes' b128 reads tile all 64 banks), pre-swizzled B-frags bf16.
// 146 KB LDS -> 2 workgroups (16 waves) per 320 KB WGP.
// ---------------------------------------------------------------------------
#define GROW 324
#define BFP  24

struct SmemB {
  float f[16][JOINT_H];          // 20480 B
  float g[64][GROW];             // 82944 B
  bf16  bfrag[3][10][32][BFP];   // 46080 B   => 149504 B total
};

__global__ __launch_bounds__(256) void rnnt_joint_wmma(
    const float* __restrict__ fws, const float* __restrict__ gws,
    const bf16* __restrict__ bws, const float* __restrict__ outb,
    float* __restrict__ dst)
{
  extern __shared__ char smem_raw[];
  SmemB* sm = (SmemB*)smem_raw;

  const int tid = threadIdx.x;
  const int bid = blockIdx.x;
  const int b   = bid >> 6;                // 32 t-blocks * 2 u-blocks per b
  const int rr  = bid & 63;
  const int t0  = (rr >> 1) * 16;
  const int u0  = (rr & 1) * 64;

  // ---- stage f / g / B-frags into LDS as raw 16B chunks (ASYNCcnt) -------
  const float* fsrc = fws + (size_t)(b * T_ + t0) * JOINT_H;  // 16 rows f32
  const float* gsrc = gws + (size_t)(b * U_ + u0) * JOINT_H;  // 64 rows f32
  for (int c = tid; c < 16 * 80; c += 256) {                  // f: 20 KB
    int row = c / 80, col = c - row * 80;
    CP16(&sm->f[row][col * 4], fsrc + row * JOINT_H + col * 4);
  }
  for (int c = tid; c < 64 * 80; c += 256) {                  // g: 80 KB
    int row = c / 80, col = c - row * 80;
    CP16(&sm->g[row][col * 4], gsrc + row * JOINT_H + col * 4);
  }
  for (int c = tid; c < 3 * 10 * 32 * 2; c += 256) {          // B-frags: 30 KB
    int lr = c >> 1, half = c & 1;                            // lane-row, 16B half
    CP16((char*)&sm->bfrag[0][0][0][0] + lr * (BFP * 2) + half * 16,
         bws + lr * 16 + half * 8);
  }
  WAIT_ASYNC();
  __syncthreads();

  const int lane = tid & 31;
  const int wv   = tid >> 5;
  const int lm   = lane & 15;
  const int hi   = lane >> 4;

  for (int it = 0; it < 8; ++it) {         // 2 t's x 4 u-tiles per wave
    const int tl = wv * 2 + (it >> 2);
    const int ut = it & 3;
    const float* grow = &sm->g[ut * 16 + lm][0];
    const float* frow = &sm->f[tl][0];

    // Build all 10 A-fragments (joint rows) once; reuse across 3 c-tiles.
    // f32 add + f32 max + packed bf16 convert (v_cvt_pk_bf16_f32).
    v16bf afrag[10];
#pragma unroll
    for (int ks = 0; ks < 10; ++ks) {
      const int b1 = ks * 32 + 8 * hi;
      f32x4 fa0 = *(const f32x4*)(frow + b1);        // broadcast across lanes
      f32x4 fa1 = *(const f32x4*)(frow + b1 + 4);
      f32x4 fb0 = *(const f32x4*)(frow + b1 + 16);
      f32x4 fb1 = *(const f32x4*)(frow + b1 + 20);
      f32x4 ga0 = *(const f32x4*)(grow + b1);        // conflict-free b128
      f32x4 ga1 = *(const f32x4*)(grow + b1 + 4);
      f32x4 gb0 = *(const f32x4*)(grow + b1 + 16);
      f32x4 gb1 = *(const f32x4*)(grow + b1 + 20);
      v16bf a;
#pragma unroll
      for (int e = 0; e < 4; ++e) {
        a[e]      = (bf16)fmaxf(fa0[e] + ga0[e], 0.0f);
        a[e + 4]  = (bf16)fmaxf(fa1[e] + ga1[e], 0.0f);
        a[e + 8]  = (bf16)fmaxf(fb0[e] + gb0[e], 0.0f);
        a[e + 12] = (bf16)fmaxf(fb1[e] + gb1[e], 0.0f);
      }
      afrag[ks] = a;
    }

#pragma unroll
    for (int ct = 0; ct < 3; ++ct) {
      v8f acc = {};
#pragma unroll
      for (int ks = 0; ks < 10; ++ks) {
        const v8bf* bp = (const v8bf*)&sm->bfrag[ct][ks][lane][0];
        v16bf bvv = __builtin_shufflevector(bp[0], bp[1],
            0, 1, 2, 3, 4, 5, 6, 7, 8, 9, 10, 11, 12, 13, 14, 15);
        acc = __builtin_amdgcn_wmma_f32_16x16x32_bf16(
            false, afrag[ks], false, bvv, (short)0, acc, false, false);
      }
      const int c = ct * 16 + lm;          // N = lane%16
      if (c < NUM_C) {                     // predication only around stores
        const float bb = outb[c];
        const int t = t0 + tl;
#pragma unroll
        for (int r = 0; r < 8; ++r) {      // M (=u) = r + 8*(lane/16)
          const int u = u0 + ut * 16 + r + 8 * hi;
          dst[(((size_t)(b * T_ + t)) * U_ + u) * NUM_C + c] = acc[r] + bb;
        }
      }
    }
  }
}

// ---------------------------------------------------------------------------
extern "C" void kernel_launch(void* const* d_in, const int* in_sizes, int n_in,
                              void* d_out, int out_size, void* d_ws, size_t ws_size,
                              hipStream_t stream) {
  (void)in_sizes; (void)n_in; (void)out_size; (void)ws_size;
  const float* enc_out  = (const float*)d_in[0];
  const float* pred_out = (const float*)d_in[1];
  const float* enc_w    = (const float*)d_in[2];
  const float* enc_b    = (const float*)d_in[3];
  const float* pred_w   = (const float*)d_in[4];
  const float* pred_b   = (const float*)d_in[5];
  const float* out_w    = (const float*)d_in[6];
  const float* out_b    = (const float*)d_in[7];
  float* dst = (float*)d_out;

  float* fws = (float*)d_ws;                         // [B*T, 320] f32
  float* gws = fws + (size_t)B_ * T_ * JOINT_H;      // [B*U, 320] f32
  bf16*  bws = (bf16*)(gws + (size_t)B_ * U_ * JOINT_H);  // [3*10*32*16] bf16

  rnnt_prep_bfrag<<<dim3(1), dim3(256), 0, stream>>>(out_w, bws);
  {
    int M = B_ * T_, K = ENC_H;                      // 4096 x 768
    int waves = (M / 16) * (JOINT_H / 16);           // 5120
    rnnt_proj_gemm<<<dim3(waves / 8), dim3(256), 0, stream>>>(
        enc_out, enc_w, enc_b, fws, M, K);
  }
  {
    int M = B_ * U_, K = PRED_H;                     // 1024 x 320
    int waves = (M / 16) * (JOINT_H / 16);           // 1280
    rnnt_proj_gemm<<<dim3(waves / 8), dim3(256), 0, stream>>>(
        pred_out, pred_w, pred_b, gws, M, K);
  }
  {
    dim3 grid(B_ * (T_ / 16) * (U_ / 64));           // 512 workgroups
    rnnt_joint_wmma<<<grid, dim3(256), sizeof(SmemB), stream>>>(
        fws, gws, bws, out_b, dst);
  }
}
